// AggregationLoss_32908039422363
// MI455X (gfx1250) — matrix operand order repaired and checked by reading, not compiled, not run on hardware
//
#include <hip/hip_runtime.h>
#include <hip/hip_bf16.h>
#include <math.h>

typedef __attribute__((ext_vector_type(16))) _Float16 v16h;
typedef __attribute__((ext_vector_type(8)))  float    v8f;
typedef __attribute__((ext_vector_type(4)))  float    f4;

#define B_   16
#define C_   4
#define NPIX 541696            /* 736*736 */
#define NSEG 33
#define TROW 8                 /* floats per table row (psum0..3, ksum, rsum, pad, pad) */
#define TSZ  (NSEG * TROW)     /* 264 floats per batch */

#if defined(__AMDGCN__) && __has_builtin(__builtin_amdgcn_global_load_async_to_lds_b32)
#define HAVE_ASYNC 1
#else
#define HAVE_ASYNC 0
#if defined(__AMDGCN__)
#warning "gfx1250 async-to-LDS builtin unavailable; using sync label staging fallback"
#endif
#endif

__device__ __forceinline__ void stage_b32(int* dst_lds, const int* src_glb) {
#if HAVE_ASYNC
  __builtin_amdgcn_global_load_async_to_lds_b32(
      (__attribute__((address_space(1))) int*)const_cast<int*>(src_glb),
      (__attribute__((address_space(3))) int*)dst_lds,
      0, 0);
#else
  *dst_lds = *src_glb;
#endif
}

#if HAVE_ASYNC
#define WAIT_ASYNC(n) asm volatile("s_wait_asynccnt %0" :: "i"(n) : "memory")
#else
#define WAIT_ASYNC(n)
#endif

/* ws layout (as float words):
   [0, B_*TSZ)   per-batch segment tables
   [B_*TSZ]      int num_kernel (atomicMax)
   [B_*TSZ + 1]  float loss accumulator                                     */

__global__ void agg_init(float* ws) {
  const int n = B_ * TSZ + 2;
  for (int i = threadIdx.x + blockIdx.x * blockDim.x; i < n;
       i += blockDim.x * gridDim.x)
    ws[i] = 0.0f;                       /* 0.0f bits == (int)0 bits */
}

__global__ __launch_bounds__(256) void agg_pass1(
    const float* __restrict__ pred, const float* __restrict__ kmask,
    const int* __restrict__ rlab, const int* __restrict__ klab,
    float* __restrict__ gtable, int* __restrict__ nk)
{
  const int b = blockIdx.y;
  const float* predb  = pred  + (size_t)b * C_ * NPIX;
  const float* kmaskb = kmask + (size_t)b * NPIX;
  const int*   klabb  = klab  + (size_t)b * NPIX;
  const int*   rlabb  = rlab  + (size_t)b * NPIX;

  __shared__ float stab[TSZ];
  __shared__ int   skl2[2][8][32];      /* double-buffered per-wave klab staging */
  __shared__ int   smax;

  const int  tid  = threadIdx.x;
  const int  lane = tid & 31;
  const int  wv   = __builtin_amdgcn_readfirstlane(tid >> 5);  /* wave-uniform */
  const int  row  = lane & 15;          /* A-matrix row / B-matrix column */
  const bool hi   = lane >= 16;
  const bool lastb = (b == B_ - 1);

  for (int i = tid; i < TSZ; i += 256) stab[i] = 0.0f;
  if (tid == 0) smax = 0;
  __syncthreads();

  v8f acc0 = {0.f, 0.f, 0.f, 0.f, 0.f, 0.f, 0.f, 0.f};   /* segs 1..16  */
  v8f acc1 = {0.f, 0.f, 0.f, 0.f, 0.f, 0.f, 0.f, 0.f};   /* segs 17..32 */
  int lmax = 0;
  const int t0 = row + 1;
  const int t1 = row + 17;
  const _Float16 oneh  = (_Float16)1.0f;
  const _Float16 zeroh = (_Float16)0.0f;

  const int nchunks = NPIX / 32;        /* exact: N % 32 == 0 */
  const int stride  = gridDim.x * 8;
  int chunk = blockIdx.x * 8 + wv;
  int par = 0;

  if (chunk < nchunks)                  /* prologue: stage first chunk's labels */
    stage_b32(&skl2[0][wv][lane], &klabb[chunk * 32 + lane]);

  for (; chunk < nchunks; chunk += stride, par ^= 1) {
    const int base = chunk * 32;
    const int nxt  = chunk + stride;
    if (nxt < nchunks)                  /* prefetch next chunk's labels */
      stage_b32(&skl2[par ^ 1][wv][lane], &klabb[nxt * 32 + lane]);

    /* own-pixel loads: rsum accumulation (kmask is pass1-only -> NT) */
    const int   rl = rlabb[base + lane];
    const float km = __builtin_nontemporal_load(&kmaskb[base + lane]);
    if (rl > 0) atomicAdd(&stab[rl * TROW + 5], km);

    /* A fragment (16x32 f16): rows 0..3 = pred channels, row 4 = kmask.
       lanes 0-15: K = 0..7 and 16..23 ; lanes 16-31: K = 8..15 and 24..31.
       pred/kmask stream once -> non-temporal loads. */
    float xs[16];
    #pragma unroll
    for (int j = 0; j < 16; ++j) xs[j] = 0.0f;
    if (row < 5) {
      const float* srow = (row == 4) ? kmaskb : (predb + (size_t)row * NPIX);
      const f4* p = (const f4*)(srow + base + (hi ? 8 : 0));
      const f4* q = (const f4*)(srow + base + 16 + (hi ? 8 : 0));
      const f4 x0 = __builtin_nontemporal_load(p);
      const f4 x1 = __builtin_nontemporal_load(p + 1);
      const f4 y0 = __builtin_nontemporal_load(q);
      const f4 y1 = __builtin_nontemporal_load(q + 1);
      xs[0]=x0.x;  xs[1]=x0.y;  xs[2]=x0.z;  xs[3]=x0.w;
      xs[4]=x1.x;  xs[5]=x1.y;  xs[6]=x1.z;  xs[7]=x1.w;
      xs[8]=y0.x;  xs[9]=y0.y;  xs[10]=y0.z; xs[11]=y0.w;
      xs[12]=y1.x; xs[13]=y1.y; xs[14]=y1.z; xs[15]=y1.w;
    }
    v16h a;
    #pragma unroll
    for (int j = 0; j < 16; ++j) a[j] = (_Float16)xs[j];

    /* wait for current chunk's async label copy (1 prefetch may stay in flight) */
    WAIT_ASYNC(1);

    /* B fragments (32x16 f16, one-hot of klab against this lane's segment):
       lanes 0-15 hold K=0..7,16..23, lanes 16-31 hold K=8..15,24..31, column N = row */
    const int* sk = &skl2[par][wv][hi ? 8 : 0];
    v16h b0, b1;
    #pragma unroll
    for (int j = 0; j < 16; ++j) {
      const int kv = sk[j < 8 ? j : j + 8];
      b0[j] = (kv == t0) ? oneh : zeroh;
      b1[j] = (kv == t1) ? oneh : zeroh;
      if (lastb) lmax = max(lmax, kv);  /* union over lanes covers all 32 pixels */
    }

    acc0 = __builtin_amdgcn_wmma_f32_16x16x32_f16(false, a, false, b0,
                                                  (short)0, acc0, false, false);
    acc1 = __builtin_amdgcn_wmma_f32_16x16x32_f16(false, a, false, b1,
                                                  (short)0, acc1, false, false);
  }
  WAIT_ASYNC(0);

  /* D layout: lanes 0-15, VGPR m = row m (rows 5..15 are zero).
     lane n column n -> segment n+1 (acc0) / n+17 (acc1). */
  if (lane < 16) {
    #pragma unroll
    for (int m = 0; m < 5; ++m) {
      atomicAdd(&stab[(lane + 1)  * TROW + m], acc0[m]);
      atomicAdd(&stab[(lane + 17) * TROW + m], acc1[m]);
    }
  }
  if (lastb) atomicMax(&smax, lmax);
  __syncthreads();

  float* gt = gtable + (size_t)b * TSZ;
  for (int i = tid; i < TSZ; i += 256) {
    const float v = stab[i];
    if (v != 0.0f) atomicAdd(&gt[i], v);
  }
  if (lastb && tid == 0) atomicMax(nk, smax);
}

__global__ __launch_bounds__(256) void agg_pass2(
    const float* __restrict__ pred, const float* __restrict__ rmask,
    const int* __restrict__ rlab, const int* __restrict__ klab,
    const float* __restrict__ gtable, float* __restrict__ lossp)
{
  const int b = blockIdx.y;
  const float* predb  = pred  + (size_t)b * C_ * NPIX;
  const float* rmaskb = rmask + (size_t)b * NPIX;
  const int*   klabb  = klab  + (size_t)b * NPIX;
  const int*   rlabb  = rlab  + (size_t)b * NPIX;

  __shared__ float stab[TSZ];
  __shared__ float sred[256];
  const int tid = threadIdx.x;

  for (int i = tid; i < TSZ; i += 256) stab[i] = gtable[(size_t)b * TSZ + i];
  __syncthreads();

  float accum = 0.0f;
  for (int p = blockIdx.x * 256 + tid; p < NPIX; p += gridDim.x * 256) {
    const int   kl = klabb[p];          /* labels RT: should hit L2 from pass1 */
    const int   rl = rlabb[p];
    const float rm = __builtin_nontemporal_load(&rmaskb[p]);
    /* table row 0 is exactly zero, so label==0 needs no masking */
    const f4    ps   = *(const f4*)&stab[kl * TROW];   /* 32B aligned */
    const float ksum = stab[kl * TROW + 4];
    const float rsum = stab[rl * TROW + 5];
    const float inv  = 1.0f / (ksum + 1.0f);

    const float p0 = __builtin_nontemporal_load(&predb[p]);
    const float p1 = __builtin_nontemporal_load(&predb[(size_t)NPIX + p]);
    const float p2 = __builtin_nontemporal_load(&predb[(size_t)2 * NPIX + p]);
    const float p3 = __builtin_nontemporal_load(&predb[(size_t)3 * NPIX + p]);
    const float d0 = p0 * rm - ps.x * inv;
    const float d1 = p1 * rm - ps.y * inv;
    const float d2 = p2 * rm - ps.z * inv;
    const float d3 = p3 * rm - ps.w * inv;
    const float s  = d0*d0 + d1*d1 + d2*d2 + d3*d3;
    const float nrm = sqrtf(s);
    const float dd  = fmaxf(nrm - 0.5f, 0.0f);
    accum += logf(fmaf(dd, dd, 1.0f)) / (rsum + 1.0f);
  }

  sred[tid] = accum;
  __syncthreads();
  for (int off = 128; off > 0; off >>= 1) {
    if (tid < off) sred[tid] += sred[tid + off];
    __syncthreads();
  }
  if (tid == 0) atomicAdd(lossp, sred[0]);
}

__global__ void agg_final(const float* lossp, const int* nk, float* out) {
  out[0] = lossp[0] / (float)nk[0];
}

extern "C" void kernel_launch(void* const* d_in, const int* in_sizes, int n_in,
                              void* d_out, int out_size, void* d_ws, size_t ws_size,
                              hipStream_t stream) {
  const float* pred  = (const float*)d_in[0];   /* [16,4,736,736] f32 */
  const float* rmask = (const float*)d_in[1];   /* [16,1,736,736] f32 */
  const float* kmask = (const float*)d_in[2];   /* [16,1,736,736] f32 */
  const int*   rlab  = (const int*)d_in[3];     /* text_mask_ndi_labels   */
  const int*   klab  = (const int*)d_in[4];     /* kernel_mask_ndi_labels */

  float* ws     = (float*)d_ws;
  float* gtable = ws;
  int*   nk     = (int*)(ws + B_ * TSZ);
  float* lossp  = ws + B_ * TSZ + 1;

  agg_init <<<8, 256, 0, stream>>>(ws);
  agg_pass1<<<dim3(64,  B_), 256, 0, stream>>>(pred, kmask, rlab, klab, gtable, nk);
  agg_pass2<<<dim3(256, B_), 256, 0, stream>>>(pred, rmask, rlab, klab, gtable, lossp);
  agg_final<<<1, 1, 0, stream>>>(lossp, nk, (float*)d_out);
}
